// MLPNet_80633716015170
// MI455X (gfx1250) — compile-verified, matrix operand order checked by reading, not run on hardware
//
#include <hip/hip_runtime.h>
#include <hip/hip_bf16.h>

typedef __attribute__((ext_vector_type(16))) _Float16 v16h;
typedef __attribute__((ext_vector_type(8)))  _Float16 v8h;
typedef __attribute__((ext_vector_type(8)))  float    v8f;

// K index held by (lane, half-slot) for 16-bit A/B WMMA fragments (ISA 7.12.2):
// lanes 0-15: s=0..7 -> K=0..7,  s=8..15 -> K=16..23
// lanes 16-31: s=0..7 -> K=8..15, s=8..15 -> K=24..31
__device__ __forceinline__ int kmap(int l, int s) {
    return (s < 8 ? s : s + 8) + ((l & 16) ? 8 : 0);
}

// ---------------------------------------------------------------------------
// Pack a row-major f32 weight matrix W[Krows][Ncols] into f16 B-fragments.
//   F[((kt*ntiles + nt)*32 + lane)*16 + s] = W[kt*32 + kmap(lane,s)][nt*16 + (lane&15)]
// ---------------------------------------------------------------------------
__global__ void pack_b_kernel(const float* __restrict__ W, _Float16* __restrict__ F,
                              int Krows, int Ncols) {
    int total = Krows * Ncols;
    int t = blockIdx.x * blockDim.x + threadIdx.x;
    if (t >= total) return;
    int s  = t & 15;
    int l  = (t >> 4) & 31;
    int f  = t >> 9;
    int ntiles = Ncols >> 4;
    int nt = f % ntiles;
    int kt = f / ntiles;
    int k  = kt * 32 + kmap(l, s);
    int n  = nt * 16 + (l & 15);
    F[t] = (_Float16)W[(size_t)k * Ncols + n];
}

__device__ __forceinline__ v16h ldB(const _Float16* F, int ntiles, int kt, int nt, int l) {
    return *(const v16h*)(F + (((size_t)(kt * ntiles + nt)) * 32 + l) * 16);
}

// Store a C-layout tile (relu optional) as f16 into an LDS tile of row width W.
__device__ __forceinline__ void stage_store(_Float16* tile, int W, const v8f* c,
                                            int nts, int l, bool relu) {
    int n0  = l & 15;
    int hi8 = (l & 16) ? 8 : 0;
    #pragma unroll
    for (int nt = 0; nt < nts; ++nt) {
        #pragma unroll
        for (int r = 0; r < 8; ++r) {
            float v = c[nt][r];
            if (relu) v = fmaxf(v, 0.f);
            tile[(r + hi8) * W + nt * 16 + n0] = (_Float16)v;
        }
    }
}

// Load ONE A fragment (ktile kt) from an LDS tile of row width W.
__device__ __forceinline__ v16h stage_loadA1(const _Float16* tile, int W, int kt, int l) {
    int m  = l & 15;
    int ko = (l & 16) ? 8 : 0;
    const _Float16* p = tile + m * W + kt * 32 + ko;
    v8h lo = *(const v8h*)p;
    v8h hi = *(const v8h*)(p + 16);
    v16h x;
    #pragma unroll
    for (int i = 0; i < 8; ++i) { x[i] = lo[i]; x[8 + i] = hi[i]; }
    return x;
}

// One 16x128 layer with K=128: flattened (kt,nt) stream, B double-buffered so
// each WMMA only waits on a load issued one iteration earlier.
__device__ __forceinline__ void layer128(const v16h* a, const _Float16* Bf,
                                         const float* bias, int l, v8f* c) {
    int n0 = l & 15;
    #pragma unroll
    for (int nt = 0; nt < 8; ++nt) {
        float bv = bias[nt * 16 + n0];
        c[nt] = (v8f){bv, bv, bv, bv, bv, bv, bv, bv};
    }
    v16h bcur = ldB(Bf, 8, 0, 0, l);
    #pragma unroll
    for (int i = 0; i < 32; ++i) {
        int kt = i >> 3, nt = i & 7;
        v16h bnext = bcur;
        if (i < 31) bnext = ldB(Bf, 8, (i + 1) >> 3, (i + 1) & 7, l);
        c[nt] = __builtin_amdgcn_wmma_f32_16x16x32_f16(
            false, a[kt], false, bcur, (short)0, c[nt], false, false);
        bcur = bnext;
    }
}

// ---------------------------------------------------------------------------
// Node kernel: x = sigmoid(r(r(r(h W0) W1) W2) Wg) * r(r(r(h W0) W1) W2)
// One wave -> 16 node rows. Output stored f16 to xh[N][128].
// ---------------------------------------------------------------------------
__global__ void __launch_bounds__(128)
node_mlp_kernel(const float* __restrict__ h,
                const float* __restrict__ b0, const float* __restrict__ b1,
                const float* __restrict__ b2, const float* __restrict__ bg,
                const _Float16* __restrict__ W0f, const _Float16* __restrict__ W1f,
                const _Float16* __restrict__ W2f, const _Float16* __restrict__ Wgf,
                _Float16* __restrict__ xh, int N) {
    __shared__ _Float16 lds[4][16 * 128];
    int l = threadIdx.x & 31;
    int w = threadIdx.x >> 5;
    _Float16* tile = lds[w];
    int row0 = (blockIdx.x * 4 + w) * 16;
    if (row0 >= N) return;

    int m  = l & 15;
    int rr = row0 + m; if (rr >= N) rr = N - 1;
    int ko = (l & 16) ? 8 : 0;

    // A fragments from h (f32 -> f16), two 8-element chunks per ktile
    v16h a[4];
    #pragma unroll
    for (int kt = 0; kt < 4; ++kt) {
        const float* p = h + (size_t)rr * 128 + kt * 32 + ko;
        v16h x;
        #pragma unroll
        for (int i = 0; i < 8; ++i) { x[i] = (_Float16)p[i]; x[8 + i] = (_Float16)p[16 + i]; }
        a[kt] = x;
    }

    v8f c[8];
    layer128(a, W0f, b0, l, c);
    stage_store(tile, 128, c, 8, l, true);
    #pragma unroll
    for (int kt = 0; kt < 4; ++kt) a[kt] = stage_loadA1(tile, 128, kt, l);

    layer128(a, W1f, b1, l, c);
    stage_store(tile, 128, c, 8, l, true);
    #pragma unroll
    for (int kt = 0; kt < 4; ++kt) a[kt] = stage_loadA1(tile, 128, kt, l);

    layer128(a, W2f, b2, l, c);
    stage_store(tile, 128, c, 8, l, true);      // act tile lives in LDS from here on
    #pragma unroll
    for (int kt = 0; kt < 4; ++kt) a[kt] = stage_loadA1(tile, 128, kt, l);

    layer128(a, Wgf, bg, l, c);                  // gate pre-activation

    int hi8 = (l & 16) ? 8 : 0;
    int n0  = l & 15;
    #pragma unroll
    for (int nt = 0; nt < 8; ++nt) {
        #pragma unroll
        for (int r = 0; r < 8; ++r) {
            int row = row0 + r + hi8;
            if (row < N) {
                float av = (float)tile[(r + hi8) * 128 + nt * 16 + n0];  // relu(x@W2)
                float g  = 1.f / (1.f + __expf(-c[nt][r]));
                xh[(size_t)row * 128 + nt * 16 + n0] = (_Float16)(g * av);
            }
        }
    }
}

// ---------------------------------------------------------------------------
// Edge kernel: he = [x[src] ; x[dst]] (16x256) -> R0(relu) -> R1(relu) -> R2
// One wave -> 16 edges. A + B both software-pipelined.
// R2 done as WMMA with N padded 2 -> 16.
// ---------------------------------------------------------------------------
__global__ void __launch_bounds__(128)
edge_mlp_kernel(const _Float16* __restrict__ xh,
                const int* __restrict__ src, const int* __restrict__ dst,
                const _Float16* __restrict__ R0f, const _Float16* __restrict__ R1f,
                const float* __restrict__ r0, const float* __restrict__ r1,
                const float* __restrict__ R2, const float* __restrict__ r2,
                float* __restrict__ out, int E) {
    __shared__ _Float16 lds[4][16 * 128];
    int l = threadIdx.x & 31;
    int w = threadIdx.x >> 5;
    _Float16* tile = lds[w];
    long e0 = ((long)blockIdx.x * 4 + w) * 16;
    if (e0 >= E) return;

    int m = l & 15;
    long e = e0 + m; if (e >= E) e = E - 1;
    int ko = (l & 16) ? 8 : 0;
    const _Float16* srow = xh + (size_t)src[e] * 128;
    const _Float16* drow = xh + (size_t)dst[e] * 128;

    int n0 = l & 15;

    // Gathered A fragment for ktile kt of he (kt<4 = src half, kt>=4 = dst half)
    auto loadA_he = [&](int kt) -> v16h {
        const _Float16* base = (kt < 4) ? srow : drow;
        const _Float16* p = base + (kt & 3) * 32 + ko;
        v8h lo = *(const v8h*)p;
        v8h hi = *(const v8h*)(p + 16);
        v16h x;
        #pragma unroll
        for (int i = 0; i < 8; ++i) { x[i] = lo[i]; x[8 + i] = hi[i]; }
        return x;
    };

    // R0: [16x256] @ [256x128]   (8 ktiles x 8 ntiles, A+B pipelined)
    v8f c[8];
    #pragma unroll
    for (int nt = 0; nt < 8; ++nt) {
        float bv = r0[nt * 16 + n0];
        c[nt] = (v8f){bv, bv, bv, bv, bv, bv, bv, bv};
    }
    {
        v16h acur = loadA_he(0);
        v16h anext = acur;
        v16h bcur = ldB(R0f, 8, 0, 0, l);
        #pragma unroll
        for (int i = 0; i < 64; ++i) {
            int kt = i >> 3, nt = i & 7;
            if (nt == 0) anext = (kt < 7) ? loadA_he(kt + 1) : acur;  // 7-iter lead
            v16h bnext = bcur;
            if (i < 63) bnext = ldB(R0f, 8, (i + 1) >> 3, (i + 1) & 7, l);
            c[nt] = __builtin_amdgcn_wmma_f32_16x16x32_f16(
                false, acur, false, bcur, (short)0, c[nt], false, false);
            bcur = bnext;
            if (nt == 7) acur = anext;
        }
    }
    stage_store(tile, 128, c, 8, l, true);

    // R1: [16x128] @ [128x64]   (4 ktiles x 4 ntiles, A from LDS, B pipelined)
    v8f c1[4];
    #pragma unroll
    for (int nt = 0; nt < 4; ++nt) {
        float bv = r1[nt * 16 + n0];
        c1[nt] = (v8f){bv, bv, bv, bv, bv, bv, bv, bv};
    }
    {
        v16h acur = stage_loadA1(tile, 128, 0, l);
        v16h anext = acur;
        v16h bcur = ldB(R1f, 4, 0, 0, l);
        #pragma unroll
        for (int i = 0; i < 16; ++i) {
            int kt = i >> 2, nt = i & 3;
            if (nt == 0) anext = (kt < 3) ? stage_loadA1(tile, 128, kt + 1, l) : acur;
            v16h bnext = bcur;
            if (i < 15) bnext = ldB(R1f, 4, (i + 1) >> 2, (i + 1) & 3, l);
            c1[nt] = __builtin_amdgcn_wmma_f32_16x16x32_f16(
                false, acur, false, bcur, (short)0, c1[nt], false, false);
            bcur = bnext;
            if (nt == 3) acur = anext;
        }
    }
    stage_store(tile, 64, c1, 4, l, true);

    // R2: [16x64] @ [64x2], N zero-padded to 16 so it stays on the matrix pipe
    float bz = (n0 < 2) ? r2[n0] : 0.f;
    v8f c2 = {bz, bz, bz, bz, bz, bz, bz, bz};
    #pragma unroll
    for (int kt = 0; kt < 2; ++kt) {
        v16h ak = stage_loadA1(tile, 64, kt, l);
        v16h b;
        #pragma unroll
        for (int s = 0; s < 16; ++s) {
            int k = kt * 32 + kmap(l, s);
            b[s] = (n0 < 2) ? (_Float16)R2[k * 2 + n0] : (_Float16)0.f;
        }
        c2 = __builtin_amdgcn_wmma_f32_16x16x32_f16(
            false, ak, false, b, (short)0, c2, false, false);
    }

    int hi8 = (l & 16) ? 8 : 0;
    if (n0 < 2) {
        #pragma unroll
        for (int r = 0; r < 8; ++r) {
            long ee = e0 + r + hi8;
            if (ee < E) out[ee * 2 + n0] = c2[r];
        }
    }
}

// ---------------------------------------------------------------------------
// Launcher. Input order: h,src,dst, W0,b0, W1,b1, W2,b2, Wg,bg, R0,r0, R1,r1, R2,r2
// ---------------------------------------------------------------------------
extern "C" void kernel_launch(void* const* d_in, const int* in_sizes, int n_in,
                              void* d_out, int out_size, void* d_ws, size_t ws_size,
                              hipStream_t stream) {
    const float* h   = (const float*)d_in[0];
    const int*   src = (const int*)d_in[1];
    const int*   dst = (const int*)d_in[2];
    const float* W0  = (const float*)d_in[3];  const float* b0 = (const float*)d_in[4];
    const float* W1  = (const float*)d_in[5];  const float* b1 = (const float*)d_in[6];
    const float* W2  = (const float*)d_in[7];  const float* b2 = (const float*)d_in[8];
    const float* Wg  = (const float*)d_in[9];  const float* bg = (const float*)d_in[10];
    const float* R0  = (const float*)d_in[11]; const float* r0 = (const float*)d_in[12];
    const float* R1  = (const float*)d_in[13]; const float* r1 = (const float*)d_in[14];
    const float* R2  = (const float*)d_in[15]; const float* r2 = (const float*)d_in[16];
    float* out = (float*)d_out;

    int N = in_sizes[0] / 128;
    int E = in_sizes[1];

    char*  ws = (char*)d_ws;
    size_t o  = 0;
    _Float16* xh  = (_Float16*)(ws + o); o += ((size_t)N * 128 * sizeof(_Float16) + 255) & ~(size_t)255;
    _Float16* W0f = (_Float16*)(ws + o); o += 128 * 128 * sizeof(_Float16);
    _Float16* W1f = (_Float16*)(ws + o); o += 128 * 128 * sizeof(_Float16);
    _Float16* W2f = (_Float16*)(ws + o); o += 128 * 128 * sizeof(_Float16);
    _Float16* Wgf = (_Float16*)(ws + o); o += 128 * 128 * sizeof(_Float16);
    _Float16* R0f = (_Float16*)(ws + o); o += 256 * 128 * sizeof(_Float16);
    _Float16* R1f = (_Float16*)(ws + o); o += 128 *  64 * sizeof(_Float16);

    // Pack weights into f16 WMMA B-fragment layout
    pack_b_kernel<<<(128 * 128 + 255) / 256, 256, 0, stream>>>(W0, W0f, 128, 128);
    pack_b_kernel<<<(128 * 128 + 255) / 256, 256, 0, stream>>>(W1, W1f, 128, 128);
    pack_b_kernel<<<(128 * 128 + 255) / 256, 256, 0, stream>>>(W2, W2f, 128, 128);
    pack_b_kernel<<<(128 * 128 + 255) / 256, 256, 0, stream>>>(Wg, Wgf, 128, 128);
    pack_b_kernel<<<(256 * 128 + 255) / 256, 256, 0, stream>>>(R0, R0f, 256, 128);
    pack_b_kernel<<<(128 *  64 + 255) / 256, 256, 0, stream>>>(R1, R1f, 128,  64);

    // Node transform: 4 waves/block, 16 rows/wave
    int nblocks = (N + 63) / 64;
    node_mlp_kernel<<<nblocks, 128, 0, stream>>>(h, b0, b1, b2, bg,
                                                 W0f, W1f, W2f, Wgf, xh, N);

    // Edge readout: 4 waves/block, 16 edges/wave
    int eblocks = (E + 63) / 64;
    edge_mlp_kernel<<<eblocks, 128, 0, stream>>>(xh, src, dst, R0f, R1f,
                                                 r0, r1, R2, r2, out, E);
}
